// SimpleGRU_89687507076001
// MI455X (gfx1250) — matrix operand order repaired and checked by reading, not compiled, not run on hardware
//
#include <hip/hip_runtime.h>
#include <hip/hip_bf16.h>

// ---------------------------------------------------------------------------
// 2-layer GRU for MI455X (gfx1250, wave32, WMMA).
//   Phase 0: weights -> bf16 pre-swizzled WMMA B-fragments (once),
//            layer input activations -> bf16 row-major.
//   Phase 1 (per layer): Gx[g] = Xbf @ Wx[g] (fp32 accum), fully parallel.
//   Phase 2 (per layer): 512 sequential steps, 2 kernels/step:
//            A: r = sig(Gx_r + h@Wr_h + br), z = sig(...); store z, (r*h) bf16
//            B: n = tanh(Gx_h + (r*h)@Wh_h + bh); h = (1-z)*n + z*h; emit y_s
// All WMMA inner loops are software-pipelined (ping-pong fragment regs) so
// fragment loads for iteration kt+1 are in flight during the WMMAs of kt.
// Gx (402 MB, stream once-write/once-read) uses non-temporal hints so the
// 25 MB weight fragments + recurrent state stay resident in the 192 MB L2.
// ---------------------------------------------------------------------------

typedef __bf16 bf16;
typedef __attribute__((ext_vector_type(16))) __bf16 v16bf;
typedef __attribute__((ext_vector_type(8)))  float  v8f;
typedef __attribute__((ext_vector_type(4)))  unsigned int u32x4;

union FragU { v16bf v; u32x4 q[2]; };

constexpr int Bb = 64, Ss = 512, Dd = 1024, Hh = 1024, Ll = 2;
constexpr int DH = Dd + Hh;            // 2048
constexpr int BS = Bb * Ss;            // 32768
constexpr int KT = Hh / 32;            // 32 k-tiles of 32
constexpr int NT = Hh / 16;            // 64 n-tiles of 16
constexpr int FRAG_ELEMS = 32 * 16;    // 512 bf16 per (jt,kt) fragment
constexpr int WF_PER_MAT = NT * KT * FRAG_ELEMS;    // 1,048,576
constexpr int WF_TOTAL   = Ll * 2 * 3 * WF_PER_MAT; // 12,582,912 bf16

// workspace layout (byte offsets, all 256-aligned)
constexpr size_t OFF_WF   = 0;
constexpr size_t OFF_XBF  = OFF_WF   + (size_t)WF_TOTAL * 2;
constexpr size_t OFF_GX   = OFF_XBF  + (size_t)BS * 1024 * 2;
constexpr size_t OFF_Y0   = OFF_GX   + (size_t)3 * BS * Hh * 4;
constexpr size_t OFF_H    = OFF_Y0   + (size_t)BS * Hh * 4;
constexpr size_t OFF_HBF  = OFF_H    + (size_t)Bb * Hh * 4;
constexpr size_t OFF_RHBF = OFF_HBF  + (size_t)Bb * Hh * 2;
constexpr size_t OFF_Z    = OFF_RHBF + (size_t)Bb * Hh * 2;

__device__ __forceinline__ int wfbase(int l, int part, int g) {
  return (((l * 2 + part) * 3) + g) * WF_PER_MAT;
}

__device__ __forceinline__ v8f wmma_bf16(v16bf a, v16bf b, v8f c) {
  return __builtin_amdgcn_wmma_f32_16x16x32_bf16(false, a, false, b,
                                                 (short)0, c, false, false);
}

// contiguous pre-swizzled B fragment: 16 bf16 (32 B) per lane
__device__ __forceinline__ v16bf load_fragB(const bf16* p) {
  FragU f;
  f.q[0] = *reinterpret_cast<const u32x4*>(p);
  f.q[1] = *reinterpret_cast<const u32x4*>(p + 8);
  return f.v;
}

// A fragment from row-major bf16 activation matrix (ldk = 1024)
// lane<16: M=lane, K = kb+{0..7,16..23}; lane>=16: M=lane-16, K = kb+{8..15,24..31}
__device__ __forceinline__ v16bf load_fragA(const bf16* act, int rowBase,
                                            int ldk, int kb, int lane) {
  int m  = rowBase + (lane & 15);
  int kh = (lane & 16) ? 8 : 0;
  const bf16* p = act + m * ldk + kb + kh;
  FragU f;
  f.q[0] = *reinterpret_cast<const u32x4*>(p);
  f.q[1] = *reinterpret_cast<const u32x4*>(p + 16);
  return f.v;
}

__device__ __forceinline__ float sigmoidf_(float x) {
  return 1.0f / (1.0f + __expf(-x));
}
__device__ __forceinline__ float tanhf_(float x) {
  float t = __expf(-2.0f * fabsf(x));
  float r = (1.0f - t) / (1.0f + t);
  return copysignf(r, x);
}

// ---------------------------------------------------------------------------
// Phase 0 kernels
// ---------------------------------------------------------------------------
__global__ void k_convert_weights(const float* __restrict__ Wr,
                                  const float* __restrict__ Wz,
                                  const float* __restrict__ Wh,
                                  bf16* __restrict__ wf) {
  int idx  = blockIdx.x * 256 + threadIdx.x;          // exact: WF_TOTAL threads
  int e    = idx & 15;
  int lane = (idx >> 4) & 31;
  int kt   = (idx >> 9) & 31;
  int jt   = (idx >> 14) & 63;
  int rest = idx >> 20;
  int g    = rest % 3;
  int part = (rest / 3) & 1;
  int l    = rest / 6;
  int n    = jt * 16 + (lane & 15);
  int k    = kt * 32 + ((lane & 16) ? 8 : 0) + ((e < 8) ? e : e + 8);
  int kg   = part ? (Dd + k) : k;
  const float* Wsrc = (g == 0) ? Wr : ((g == 1) ? Wz : Wh);
  wf[idx] = (bf16)__builtin_nontemporal_load(&Wsrc[((size_t)l * DH + kg) * Hh + n]);
}

__global__ void k_convert_act(const float* __restrict__ src,
                              bf16* __restrict__ dst) {
  int i = blockIdx.x * 256 + threadIdx.x;             // BS*1024 threads exact
  dst[i] = (bf16)__builtin_nontemporal_load(&src[i]);
}

__global__ void k_zero_h(float* __restrict__ h, bf16* __restrict__ hbf) {
  int i = blockIdx.x * 256 + threadIdx.x;             // 65536 exact
  h[i] = 0.0f;
  hbf[i] = (bf16)0.0f;
}

// ---------------------------------------------------------------------------
// Phase 1: Gx[g][row][n] = sum_k Xbf[row][k] * Wx[g][k][n]
// wave computes 16 rows x 64 cols (4 fragments); 98304 waves; pipelined.
// ---------------------------------------------------------------------------
__global__ void __launch_bounds__(128)
k_gemm_gx(const bf16* __restrict__ xbf, const bf16* __restrict__ wf,
          float* __restrict__ gx, int l) {
  int lane  = threadIdx.x & 31;
  int w     = blockIdx.x * 4 + (threadIdx.x >> 5);
  int mtile = w % 2048;
  int nb    = w / 2048;        // 0..47
  int g     = nb >> 4;         // 0..2
  int jbase = (nb & 15) * 4;   // gate-local n-tile base
  int rowb  = mtile * 16;

  const bf16* wb = wf + wfbase(l, 0, g) + lane * 16;
  auto bptr = [&](int jj, int kt) {
    return wb + ((jbase + jj) * 32 + kt) * FRAG_ELEMS;
  };

  v8f c0 = {}, c1 = {}, c2 = {}, c3 = {};

  // stage 0
  v16bf aP = load_fragA(xbf, rowb, 1024, 0, lane);
  v16bf bP0 = load_fragB(bptr(0, 0));
  v16bf bP1 = load_fragB(bptr(1, 0));
  v16bf bP2 = load_fragB(bptr(2, 0));
  v16bf bP3 = load_fragB(bptr(3, 0));

  for (int kt = 0; kt < KT; kt += 2) {
    // prefetch kt+1 while computing kt
    v16bf aQ = load_fragA(xbf, rowb, 1024, (kt + 1) * 32, lane);
    v16bf bQ0 = load_fragB(bptr(0, kt + 1));
    v16bf bQ1 = load_fragB(bptr(1, kt + 1));
    v16bf bQ2 = load_fragB(bptr(2, kt + 1));
    v16bf bQ3 = load_fragB(bptr(3, kt + 1));
    c0 = wmma_bf16(aP, bP0, c0);
    c1 = wmma_bf16(aP, bP1, c1);
    c2 = wmma_bf16(aP, bP2, c2);
    c3 = wmma_bf16(aP, bP3, c3);

    // prefetch kt+2 while computing kt+1 (last round re-fetches kt: harmless)
    int kn = (kt + 2 < KT) ? kt + 2 : kt;
    aP = load_fragA(xbf, rowb, 1024, kn * 32, lane);
    bP0 = load_fragB(bptr(0, kn));
    bP1 = load_fragB(bptr(1, kn));
    bP2 = load_fragB(bptr(2, kn));
    bP3 = load_fragB(bptr(3, kn));
    c0 = wmma_bf16(aQ, bQ0, c0);
    c1 = wmma_bf16(aQ, bQ1, c1);
    c2 = wmma_bf16(aQ, bQ2, c2);
    c3 = wmma_bf16(aQ, bQ3, c3);
  }

#pragma unroll
  for (int v = 0; v < 8; ++v) {
    int row = rowb + v + ((lane & 16) ? 8 : 0);
    float* out = gx + ((size_t)g * BS + row) * Hh + (lane & 15);
    __builtin_nontemporal_store(c0[v], &out[(jbase + 0) * 16]);
    __builtin_nontemporal_store(c1[v], &out[(jbase + 1) * 16]);
    __builtin_nontemporal_store(c2[v], &out[(jbase + 2) * 16]);
    __builtin_nontemporal_store(c3[v], &out[(jbase + 3) * 16]);
  }
}

// ---------------------------------------------------------------------------
// Phase 2 kernel A: r,z gates. wave = 16 rows x 32 cols; 256 waves; pipelined.
// ---------------------------------------------------------------------------
__global__ void __launch_bounds__(128)
k_step_gates(const bf16* __restrict__ hbf, const float* __restrict__ h,
             const bf16* __restrict__ wf, const float* __restrict__ gx,
             const float* __restrict__ br, const float* __restrict__ bz,
             bf16* __restrict__ rhbf, float* __restrict__ zbuf,
             int s, int l) {
  int lane  = threadIdx.x & 31;
  int w     = blockIdx.x * 4 + (threadIdx.x >> 5);
  int mtile = w & 3;
  int nblk  = w >> 2;                 // 0..63 over 2048 cols
  int nglob = nblk * 32;
  int g     = nglob >> 10;            // 0 = r, 1 = z
  int jbase = (nglob & 1023) >> 4;    // gate-local n-tile
  int rowb  = mtile * 16;

  const bf16* wb = wf + wfbase(l, 1, g) + lane * 16;
  auto bptr = [&](int jj, int kt) {
    return wb + ((jbase + jj) * 32 + kt) * FRAG_ELEMS;
  };

  v8f c0 = {}, c1 = {};
  v16bf aP = load_fragA(hbf, rowb, 1024, 0, lane);
  v16bf bP0 = load_fragB(bptr(0, 0));
  v16bf bP1 = load_fragB(bptr(1, 0));
  for (int kt = 0; kt < KT; kt += 2) {
    v16bf aQ = load_fragA(hbf, rowb, 1024, (kt + 1) * 32, lane);
    v16bf bQ0 = load_fragB(bptr(0, kt + 1));
    v16bf bQ1 = load_fragB(bptr(1, kt + 1));
    c0 = wmma_bf16(aP, bP0, c0);
    c1 = wmma_bf16(aP, bP1, c1);
    int kn = (kt + 2 < KT) ? kt + 2 : kt;
    aP = load_fragA(hbf, rowb, 1024, kn * 32, lane);
    bP0 = load_fragB(bptr(0, kn));
    bP1 = load_fragB(bptr(1, kn));
    c0 = wmma_bf16(aQ, bQ0, c0);
    c1 = wmma_bf16(aQ, bQ1, c1);
  }

  const float* bias = (g == 0) ? br : bz;
#pragma unroll
  for (int v = 0; v < 8; ++v) {
    int m = rowb + v + ((lane & 16) ? 8 : 0);         // batch row, 0..63
#pragma unroll
    for (int jj = 0; jj < 2; ++jj) {
      int nl   = (jbase + jj) * 16 + (lane & 15);     // gate-local col
      float acc = (jj ? c1[v] : c0[v]);
      acc += __builtin_nontemporal_load(
                 &gx[((size_t)g * BS + (size_t)m * Ss + s) * Hh + nl]) +
             bias[nl];
      float val = sigmoidf_(acc);
      if (g == 0) {
        float rh = val * h[m * Hh + nl];
        rhbf[m * Hh + nl] = (bf16)rh;
      } else {
        zbuf[m * Hh + nl] = val;
      }
    }
  }
}

// ---------------------------------------------------------------------------
// Phase 2 kernel B: n gate + state update. 128 waves; pipelined.
// ---------------------------------------------------------------------------
__global__ void __launch_bounds__(128)
k_step_update(const bf16* __restrict__ rhbf, const bf16* __restrict__ wf,
              const float* __restrict__ gx, const float* __restrict__ bh,
              const float* __restrict__ zbuf, float* __restrict__ h,
              bf16* __restrict__ hbf, float* __restrict__ yout,
              int s, int l) {
  int lane  = threadIdx.x & 31;
  int w     = blockIdx.x * 4 + (threadIdx.x >> 5);
  int mtile = w & 3;
  int nblk  = w >> 2;        // 0..31 over 1024 cols
  int jbase = nblk * 2;
  int rowb  = mtile * 16;

  const bf16* wb = wf + wfbase(l, 1, 2) + lane * 16;
  auto bptr = [&](int jj, int kt) {
    return wb + ((jbase + jj) * 32 + kt) * FRAG_ELEMS;
  };

  v8f c0 = {}, c1 = {};
  v16bf aP = load_fragA(rhbf, rowb, 1024, 0, lane);
  v16bf bP0 = load_fragB(bptr(0, 0));
  v16bf bP1 = load_fragB(bptr(1, 0));
  for (int kt = 0; kt < KT; kt += 2) {
    v16bf aQ = load_fragA(rhbf, rowb, 1024, (kt + 1) * 32, lane);
    v16bf bQ0 = load_fragB(bptr(0, kt + 1));
    v16bf bQ1 = load_fragB(bptr(1, kt + 1));
    c0 = wmma_bf16(aP, bP0, c0);
    c1 = wmma_bf16(aP, bP1, c1);
    int kn = (kt + 2 < KT) ? kt + 2 : kt;
    aP = load_fragA(rhbf, rowb, 1024, kn * 32, lane);
    bP0 = load_fragB(bptr(0, kn));
    bP1 = load_fragB(bptr(1, kn));
    c0 = wmma_bf16(aQ, bQ0, c0);
    c1 = wmma_bf16(aQ, bQ1, c1);
  }

#pragma unroll
  for (int v = 0; v < 8; ++v) {
    int m = rowb + v + ((lane & 16) ? 8 : 0);
#pragma unroll
    for (int jj = 0; jj < 2; ++jj) {
      int nl   = (jbase + jj) * 16 + (lane & 15);
      float acc = (jj ? c1[v] : c0[v]);
      acc += __builtin_nontemporal_load(
                 &gx[((size_t)2 * BS + (size_t)m * Ss + s) * Hh + nl]) +
             bh[nl];
      float nv = tanhf_(acc);
      float zz = zbuf[m * Hh + nl];
      float ho = h[m * Hh + nl];
      float hn = (1.0f - zz) * nv + zz * ho;
      h[m * Hh + nl]   = hn;
      hbf[m * Hh + nl] = (bf16)hn;
      __builtin_nontemporal_store(hn, &yout[((size_t)m * Ss + s) * Hh + nl]);
    }
  }
}

// ---------------------------------------------------------------------------
extern "C" void kernel_launch(void* const* d_in, const int* in_sizes, int n_in,
                              void* d_out, int out_size, void* d_ws, size_t ws_size,
                              hipStream_t stream) {
  const float* x  = (const float*)d_in[0];
  const float* Wr = (const float*)d_in[1];
  const float* Wz = (const float*)d_in[2];
  const float* Wh = (const float*)d_in[3];
  const float* br = (const float*)d_in[4];
  const float* bz = (const float*)d_in[5];
  const float* bh = (const float*)d_in[6];
  float* out = (float*)d_out;

  char* ws = (char*)d_ws;
  bf16*  wf   = (bf16*)(ws + OFF_WF);
  bf16*  xbf  = (bf16*)(ws + OFF_XBF);
  float* gx   = (float*)(ws + OFF_GX);
  float* y0   = (float*)(ws + OFF_Y0);
  float* h    = (float*)(ws + OFF_H);
  bf16*  hbf  = (bf16*)(ws + OFF_HBF);
  bf16*  rhbf = (bf16*)(ws + OFF_RHBF);
  float* zbuf = (float*)(ws + OFF_Z);

  // Phase 0: weights -> pre-swizzled bf16 fragments (once per launch)
  k_convert_weights<<<WF_TOTAL / 256, 256, 0, stream>>>(Wr, Wz, Wh, wf);

  for (int l = 0; l < Ll; ++l) {
    // layer input -> bf16 (layer 0: x, layer 1: previous layer's output)
    const float* act_in = (l == 0) ? x : y0;
    k_convert_act<<<(BS * 1024) / 256, 256, 0, stream>>>(act_in, xbf);

    // Phase 1: input-gate contributions for all timesteps (big GEMM)
    k_gemm_gx<<<(2048 * 48) / 4, 128, 0, stream>>>(xbf, wf, gx, l);

    // h = 0
    k_zero_h<<<(Bb * Hh) / 256, 256, 0, stream>>>(h, hbf);

    const float* brl = br + l * Hh;
    const float* bzl = bz + l * Hh;
    const float* bhl = bh + l * Hh;
    float* yout = (l == Ll - 1) ? out : y0;

    // Phase 2: sequential scan
    for (int s = 0; s < Ss; ++s) {
      k_step_gates<<<64, 128, 0, stream>>>(hbf, h, wf, gx, brl, bzl,
                                           rhbf, zbuf, s, l);
      k_step_update<<<32, 128, 0, stream>>>(rhbf, wf, gx, bhl, zbuf,
                                            h, hbf, yout, s, l);
    }
  }
}
